// RGCN_18064632447204
// MI455X (gfx1250) — matrix operand order repaired and checked by reading, not compiled, not run on hardware
//
#include <hip/hip_runtime.h>
#include <hip/hip_bf16.h>

#define NN      100000
#define NE      1600000
#define NGRAPH  512
#define NREL    4
#define NTYPES  13
#define HID     128

typedef __attribute__((ext_vector_type(16))) _Float16 v16h;
typedef __attribute__((ext_vector_type(8)))  _Float16 v8h;
typedef __attribute__((ext_vector_type(8)))  float    v8f;

// ---------------------------------------------------------------------------
// Zero fill
// ---------------------------------------------------------------------------
__global__ void k_zero(float* __restrict__ p, int n) {
    int i = blockIdx.x * blockDim.x + threadIdx.x;
    if (i < n) p[i] = 0.0f;
}

// ---------------------------------------------------------------------------
// Embedding lookup: h = emb[argmax(x)], also produce f16 copy for WMMA A.
// One block (128 threads) per node.
// ---------------------------------------------------------------------------
__global__ void k_embed(const float* __restrict__ x, const float* __restrict__ emb,
                        float* __restrict__ h32, _Float16* __restrict__ h16) {
    const int row = blockIdx.x;
    const int c   = threadIdx.x;
    // argmax over 13 one-hot entries (first max wins, like jnp.argmax)
    int best = 0;
    float bv = x[(size_t)row * NTYPES];
    #pragma unroll
    for (int t = 1; t < NTYPES; ++t) {
        float v = x[(size_t)row * NTYPES + t];
        if (v > bv) { bv = v; best = t; }
    }
    float val = emb[best * HID + c];
    h32[(size_t)row * HID + c] = val;
    h16[(size_t)row * HID + c] = (_Float16)val;
}

// ---------------------------------------------------------------------------
// Convert layer weights to f16, stored TRANSPOSED: WT[m][n*128+k] = W_m[k][n]
// m = 0 -> Wroot, m = 1..4 -> Wrel[r]. 5*128*128 elements.
// ---------------------------------------------------------------------------
__global__ void k_convert_wt(const float* __restrict__ Wroot,
                             const float* __restrict__ Wrel,
                             _Float16* __restrict__ WT) {
    int idx = blockIdx.x * blockDim.x + threadIdx.x;   // < 5*16384
    int m   = idx >> 14;
    int rem = idx & 16383;
    int n   = rem >> 7;
    int k   = rem & 127;
    const float* W = (m == 0) ? Wroot : (Wrel + (size_t)(m - 1) * HID * HID);
    WT[idx] = (_Float16)W[k * HID + n];
}

// ---------------------------------------------------------------------------
// WMMA GEMM:  C[nrows x 128] = A(f16, row-major) @ W  (+ bias)
// W supplied pre-transposed as BT[n*128+k] (f16).
// Block = 128 threads = 4 waves; each wave computes one 16x16 tile.
// grid = (nrows/16, 2): wave w handles col tile blockIdx.y*4 + w.
// K=128 accumulated with 4x v_wmma_f32_16x16x32_f16.
// ---------------------------------------------------------------------------
__global__ void __launch_bounds__(128)
k_gemm128_wmma(const _Float16* __restrict__ A,
               const _Float16* __restrict__ BT,
               const float*    __restrict__ bias,
               float*          __restrict__ C,
               int use_bias) {
    const int lane  = threadIdx.x & 31;
    const int wave  = threadIdx.x >> 5;
    const int half  = lane >> 4;       // 0: lanes 0-15, 1: lanes 16-31
    const int m     = lane & 15;
    const int tileM = blockIdx.x;
    const int tileN = blockIdx.y * 4 + wave;

    const int row  = tileM * 16 + m;       // A row owned by this lane
    const int coln = tileN * 16 + m;       // B/C column owned by this lane

    const _Float16* arow = A  + (size_t)row  * HID;
    const _Float16* bcol = BT + (size_t)coln * HID;

    v8f acc = {};
    #pragma unroll
    for (int kc = 0; kc < 4; ++kc) {
        const int kb = kc * 32;
        // A in contiguous k order (same as B): lane half h holds K = kb+16h..kb+16h+7 (elems 0-7)
        // and K = kb+16h+8..kb+16h+15 (elems 8-15): two contiguous b128 loads.
        v8h alo = *(const v8h*)(arow + kb + half * 16);
        v8h ahi = *(const v8h*)(arow + kb + half * 16 + 8);
        v16h a  = __builtin_shufflevector(alo, ahi,
                    0, 1, 2, 3, 4, 5, 6, 7, 8, 9, 10, 11, 12, 13, 14, 15);
        // B in the same contiguous k order: lane half h holds K = kb+16h .. kb+16h+15 for col n:
        // one contiguous 32-byte load from the transposed weight.
        v16h b  = *(const v16h*)(bcol + kb + half * 16);
        acc = __builtin_amdgcn_wmma_f32_16x16x32_f16(
                  false, a, false, b, (short)0, acc, false, false);
    }

    const float bv = use_bias ? bias[coln] : 0.0f;
    // C/D layout: acc[r] -> row tileM*16 + half*8 + r, col coln
    float* crow = C + (size_t)(tileM * 16 + half * 8) * HID + coln;
    #pragma unroll
    for (int r = 0; r < 8; ++r)
        crow[(size_t)r * HID] = acc[r] + bv;
}

// ---------------------------------------------------------------------------
// Per-relation edge scatter: num[dst] += hr[src] (for edges of type `rel`),
// cnt[dst] += 1.  One block (128 threads) per edge; coalesced 512B rows.
// ---------------------------------------------------------------------------
__global__ void k_scatter(const float* __restrict__ hr,
                          const int* __restrict__ src,
                          const int* __restrict__ dst,
                          const int* __restrict__ etype,
                          int rel,
                          float* __restrict__ num,
                          float* __restrict__ cnt) {
    const int e = blockIdx.x;
    if (etype[e] != rel) return;
    const int s = src[e];
    const int d = dst[e];
    atomicAdd(&num[(size_t)d * HID + threadIdx.x], hr[(size_t)s * HID + threadIdx.x]);
    if (threadIdx.x == 0) atomicAdd(&cnt[d], 1.0f);
}

// ---------------------------------------------------------------------------
// out += num / max(cnt,1)  (elementwise over N*128)
// ---------------------------------------------------------------------------
__global__ void k_combine(float* __restrict__ out, const float* __restrict__ num,
                          const float* __restrict__ cnt, int total) {
    int i = blockIdx.x * blockDim.x + threadIdx.x;
    if (i >= total) return;
    float c = cnt[i >> 7];
    out[i] += num[i] / fmaxf(c, 1.0f);
}

// ---------------------------------------------------------------------------
// BatchNorm statistics: buf[0:128]=sum, buf[128:256]=sumsq (atomic reduce).
// block 256 = 2 sub-rows x 128 channels; grid-stride over rows.
// ---------------------------------------------------------------------------
__global__ void k_bn_stats(const float* __restrict__ h, float* __restrict__ buf, int n) {
    const int c   = threadIdx.x & 127;
    const int sub = threadIdx.x >> 7;
    float s = 0.0f, q = 0.0f;
    for (int row = blockIdx.x * 2 + sub; row < n; row += gridDim.x * 2) {
        float v = h[(size_t)row * HID + c];
        s += v; q += v * v;
    }
    atomicAdd(&buf[c],       s);
    atomicAdd(&buf[128 + c], q);
}

// buf[256:384]=scale, buf[384:512]=shift.  1 block, 128 threads.
__global__ void k_bn_finalize(float* __restrict__ buf,
                              const float* __restrict__ gamma,
                              const float* __restrict__ beta,
                              float invn) {
    const int c = threadIdx.x;
    float mu  = buf[c] * invn;
    float var = buf[128 + c] * invn - mu * mu;
    float sc  = gamma[c] * rsqrtf(var + 1e-5f);
    buf[256 + c] = sc;
    buf[384 + c] = beta[c] - mu * sc;
}

// ---------------------------------------------------------------------------
// Fused BN apply + PReLU + L2 row normalize; writes f32 and f16 h.
// One block (128 threads) per node row.
// ---------------------------------------------------------------------------
__global__ void k_postproc(const float* __restrict__ in, const float* __restrict__ buf,
                           const float* __restrict__ prelu_a,
                           float* __restrict__ h32, _Float16* __restrict__ h16) {
    __shared__ float red[HID];
    const int row = blockIdx.x;
    const int c   = threadIdx.x;
    float v = in[(size_t)row * HID + c] * buf[256 + c] + buf[384 + c];
    const float a = prelu_a[0];
    v = (v >= 0.0f) ? v : a * v;
    red[c] = v * v;
    __syncthreads();
    for (int s = 64; s > 0; s >>= 1) {
        if (c < s) red[c] += red[c + s];
        __syncthreads();
    }
    float inv = 1.0f / fmaxf(sqrtf(red[0]), 1e-12f);
    float o = v * inv;
    h32[(size_t)row * HID + c] = o;
    h16[(size_t)row * HID + c] = (_Float16)o;
}

// ---------------------------------------------------------------------------
// Global mean pool (accumulate) + divide
// ---------------------------------------------------------------------------
__global__ void k_pool(const float* __restrict__ h, const int* __restrict__ batch,
                       float* __restrict__ pooled, float* __restrict__ gcnt) {
    const int row = blockIdx.x;
    const int c   = threadIdx.x;
    const int g   = batch[row];
    atomicAdd(&pooled[(size_t)g * HID + c], h[(size_t)row * HID + c]);
    if (c == 0) atomicAdd(&gcnt[g], 1.0f);
}

__global__ void k_pool_div(float* __restrict__ pooled, const float* __restrict__ gcnt) {
    const int g = blockIdx.x;
    const int c = threadIdx.x;
    pooled[(size_t)g * HID + c] /= fmaxf(gcnt[g], 1.0f);
}

// ---------------------------------------------------------------------------
// Tiny MLP head layer (dense + optional relu)
// ---------------------------------------------------------------------------
__global__ void k_fc(const float* __restrict__ in, const float* __restrict__ w,
                     const float* __restrict__ b, float* __restrict__ out,
                     int rows, int din, int dout, int relu) {
    int idx = blockIdx.x * blockDim.x + threadIdx.x;
    if (idx >= rows * dout) return;
    int r = idx / dout, c = idx % dout;
    float acc = b[c];
    for (int k = 0; k < din; ++k)
        acc += in[(size_t)r * din + k] * w[(size_t)k * dout + c];
    out[idx] = relu ? fmaxf(acc, 0.0f) : acc;
}

// ---------------------------------------------------------------------------
// Launch
// ---------------------------------------------------------------------------
extern "C" void kernel_launch(void* const* d_in, const int* in_sizes, int n_in,
                              void* d_out, int out_size, void* d_ws, size_t ws_size,
                              hipStream_t stream) {
    const float* x     = (const float*)d_in[0];
    const int*   eidx  = (const int*)  d_in[1];
    const int*   etype = (const int*)  d_in[2];
    const int*   batch = (const int*)  d_in[3];
    const float* emb   = (const float*)d_in[4];
    const float* Wrel  = (const float*)d_in[5];   // [2,4,128,128]
    const float* Wroot = (const float*)d_in[6];   // [2,128,128]
    const float* cbias = (const float*)d_in[7];   // [2,128]
    const float* gamma = (const float*)d_in[8];
    const float* beta  = (const float*)d_in[9];
    const float* prelu = (const float*)d_in[10];  // [2]
    const float* fc1w  = (const float*)d_in[11];
    const float* fc1b  = (const float*)d_in[12];
    const float* fc2w  = (const float*)d_in[13];
    const float* fc2b  = (const float*)d_in[14];
    const float* outw  = (const float*)d_in[15];
    const float* outb  = (const float*)d_in[16];

    const int* src = eidx;
    const int* dst = eidx + NE;

    // workspace layout (256B aligned slices)
    char*  ws  = (char*)d_ws;
    size_t off = 0;
    auto alloc = [&](size_t bytes) -> char* {
        char* p = ws + off;
        off += (bytes + 255) & ~(size_t)255;
        return p;
    };
    float*    h32    = (float*)   alloc((size_t)NN * HID * 4);
    _Float16* h16    = (_Float16*)alloc((size_t)NN * HID * 2);
    float*    out32  = (float*)   alloc((size_t)NN * HID * 4);
    float*    hr32   = (float*)   alloc((size_t)NN * HID * 4);
    float*    num32  = (float*)   alloc((size_t)NN * HID * 4);
    float*    cnt    = (float*)   alloc((size_t)NN * 4);
    _Float16* WT     = (_Float16*)alloc((size_t)5 * HID * HID * 2);
    float*    bnbuf  = (float*)   alloc(512 * 4);
    float*    pooled = (float*)   alloc((size_t)NGRAPH * HID * 4);
    float*    gcnt   = (float*)   alloc(NGRAPH * 4);
    float*    g1     = (float*)   alloc((size_t)NGRAPH * 256 * 4);
    float*    g2     = (float*)   alloc((size_t)NGRAPH * HID * 4);

    const int total = NN * HID;

    // h0 = emb[argmax(x)]
    k_embed<<<NN, HID, 0, stream>>>(x, emb, h32, h16);

    for (int l = 0; l < 2; ++l) {
        // transpose+convert this layer's 5 weight matrices to f16
        k_convert_wt<<<(5 * HID * HID) / 256, 256, 0, stream>>>(
            Wroot + (size_t)l * HID * HID,
            Wrel  + (size_t)l * NREL * HID * HID, WT);

        // root transform: out = h @ Wroot + bias   (WMMA)
        k_gemm128_wmma<<<dim3(NN / 16, 2), 128, 0, stream>>>(
            h16, WT, cbias + l * HID, out32, 1);

        for (int r = 0; r < NREL; ++r) {
            k_zero<<<(total + 255) / 256, 256, 0, stream>>>(num32, total);
            k_zero<<<(NN + 255) / 256, 256, 0, stream>>>(cnt, NN);
            // hr = h @ Wrel[r]   (WMMA)
            k_gemm128_wmma<<<dim3(NN / 16, 2), 128, 0, stream>>>(
                h16, WT + (size_t)(1 + r) * HID * HID, cbias, hr32, 0);
            k_scatter<<<NE, HID, 0, stream>>>(hr32, src, dst, etype, r, num32, cnt);
            k_combine<<<(total + 255) / 256, 256, 0, stream>>>(out32, num32, cnt, total);
        }

        // BatchNorm + PReLU + L2 normalize
        k_zero<<<1, 256, 0, stream>>>(bnbuf, 256);
        k_bn_stats<<<256, 256, 0, stream>>>(out32, bnbuf, NN);
        k_bn_finalize<<<1, HID, 0, stream>>>(bnbuf, gamma + l * HID, beta + l * HID,
                                             1.0f / (float)NN);
        k_postproc<<<NN, HID, 0, stream>>>(out32, bnbuf, prelu + l, h32, h16);
    }

    // global mean pool
    k_zero<<<(NGRAPH * HID + 255) / 256, 256, 0, stream>>>(pooled, NGRAPH * HID);
    k_zero<<<(NGRAPH + 255) / 256, 256, 0, stream>>>(gcnt, NGRAPH);
    k_pool<<<NN, HID, 0, stream>>>(h32, batch, pooled, gcnt);
    k_pool_div<<<NGRAPH, HID, 0, stream>>>(pooled, gcnt);

    // MLP head
    k_fc<<<(NGRAPH * 256 + 255) / 256, 256, 0, stream>>>(pooled, fc1w, fc1b, g1,
                                                         NGRAPH, HID, 256, 1);
    k_fc<<<(NGRAPH * HID + 255) / 256, 256, 0, stream>>>(g1, fc2w, fc2b, g2,
                                                         NGRAPH, 256, HID, 1);
    k_fc<<<(NGRAPH + 255) / 256, 256, 0, stream>>>(g2, outw, outb, (float*)d_out,
                                                   NGRAPH, HID, 1, 0);
}
